// _GraphAttention_65481071402429
// MI455X (gfx1250) — matrix operand order repaired and checked by reading, not compiled, not run on hardware
//
#include <hip/hip_runtime.h>
#include <stdint.h>

// Problem constants (match the reference).
#define DD 1024
#define LL 1024
#define BB 4
#define HH 16
#define HD 64
#define NROWS (BB * LL)     // 4096 token rows
#define THREE_D (3 * DD)    // 3072

typedef __attribute__((ext_vector_type(16))) __bf16 v16bf;
typedef __attribute__((ext_vector_type(8)))  __bf16 v8bf;
typedef __attribute__((ext_vector_type(8)))  float  v8f;

union V16U { v16bf v; v8bf h[2]; };

__device__ __forceinline__ unsigned short f2bf_bits(float f) {
  union { float f; uint32_t u; } a; a.f = f;
  uint32_t u = a.u;
  uint32_t r = u + 0x7fffu + ((u >> 16) & 1u);     // round-to-nearest-even
  if ((u & 0x7f800000u) == 0x7f800000u) r = u;     // inf/nan passthrough
  return (unsigned short)(r >> 16);
}

// ---- WMMA operand loaders (contiguous-k layouts, wave32) -------------------
// A operand: 16x32 bf16 tile (M x K). Lane l holds row M=l&15,
// K = (l>>4)*16 .. +15 contiguous (two 8-element loads).
__device__ __forceinline__ v16bf load_a16x32(const unsigned short* src,
                                             int row_base, int k0, int ld, int lane) {
  int m = lane & 15, half = lane >> 4;
  const unsigned short* p = src + (size_t)(row_base + m) * ld + k0 + half * 16;
  V16U r;
  r.h[0] = *reinterpret_cast<const v8bf*>(p);
  r.h[1] = *reinterpret_cast<const v8bf*>(p + 8);
  return r.v;
}

// B operand: 32x16 bf16 tile (K x N) where column n is contiguous in memory
// (src row = output column, stride ld). Lane l holds column N=l&15,
// K = (l>>4)*16 .. +15 contiguous -> one 32-byte load.
__device__ __forceinline__ v16bf load_b32x16(const unsigned short* src,
                                             int col_base, int k0, int ld, int lane) {
  int n = lane & 15, half = lane >> 4;
  const unsigned short* p = src + (size_t)(col_base + n) * ld + k0 + half * 16;
  return *reinterpret_cast<const v16bf*>(p);
}

__device__ __forceinline__ v8f wmma_bf16(v16bf a, v16bf b, v8f c) {
  return __builtin_amdgcn_wmma_f32_16x16x32_bf16(false, a, false, b,
                                                 (short)0, c, false, false);
}

// ---- Small prep kernels -----------------------------------------------------
__global__ void k_clear_u32(uint32_t* p, int n) {
  int i = blockIdx.x * blockDim.x + threadIdx.x;
  if (i < n) p[i] = 0u;
}

__global__ void k_scatter_edges(const int* __restrict__ edges,
                                uint32_t* __restrict__ adj, int ne) {
  int i = blockIdx.x * blockDim.x + threadIdx.x;
  if (i >= ne) return;
  int a = edges[2 * i], b = edges[2 * i + 1];
  atomicOr(&adj[(size_t)a * 32 + (b >> 5)], 1u << (b & 31));
  atomicOr(&adj[(size_t)b * 32 + (a >> 5)], 1u << (a & 31));
}

__global__ void k_f32_to_bf16(const float* __restrict__ s,
                              unsigned short* __restrict__ d, int n) {
  int i = blockIdx.x * blockDim.x + threadIdx.x;
  if (i < n) d[i] = f2bf_bits(s[i]);
}

// ---- Fully-unrolled 16x64-tile GEMM, 3-deep K pipeline ----------------------
// C[16 rows x 64 cols] += A[16 x 1024] * W[64 cols x 1024]^T, bf16 in, fp32 acc.
// sched_barrier(0) at each step end pins program order so the machine
// scheduler cannot sink refill loads down to their consuming WMMA: the refill
// for step s+3 issues before step s+1's WMMAs, keeping ~2 load-groups in
// flight and making pre-WMMA waits partial by construction.
#define KSTAGES 3
__device__ __forceinline__ void gemm_k_pipeline(
    const unsigned short* __restrict__ amat, int row_base,
    const unsigned short* __restrict__ wmat, int col_base,
    int nloc, int lane, v8f (&acc)[4]) {
  v16bf abuf[KSTAGES];
  v16bf bbuf[KSTAGES][4];
#pragma unroll
  for (int s = 0; s < KSTAGES; ++s) {        // prologue: stages 0..2 in flight
    abuf[s] = load_a16x32(amat, row_base, s * 32, DD, lane);
#pragma unroll
    for (int t = 0; t < 4; ++t)
      bbuf[s][t] = load_b32x16(wmat, col_base + t * 16, s * 32, DD, lane);
  }
  __builtin_amdgcn_sched_barrier(0);         // prologue issues up front

#pragma unroll
  for (int s = 0; s < DD / 32; ++s) {
    const int st = s % KSTAGES;              // compile-time constant
    if ((s & 7) == 0 && (s + 8) * 32 < DD) { // folds at compile time
      __builtin_prefetch(amat + (size_t)(row_base + nloc) * DD + (s + 8) * 32, 0, 1);
      __builtin_prefetch(wmat + (size_t)(col_base + nloc) * DD + (s + 8) * 32, 0, 1);
    }
#pragma unroll
    for (int t = 0; t < 4; ++t) acc[t] = wmma_bf16(abuf[st], bbuf[st][t], acc[t]);

    const int kn = (s + KSTAGES) * 32;       // refill consumed stage, 3 ahead
    if (kn < DD) {                           // folds on final steps
      abuf[st] = load_a16x32(amat, row_base, kn, DD, lane);
#pragma unroll
      for (int t = 0; t < 4; ++t)
        bbuf[st][t] = load_b32x16(wmat, col_base + t * 16, kn, DD, lane);
    }
    __builtin_amdgcn_sched_barrier(0);       // refills may not sink past here
  }
}

// ---- QKV projection: [4096,1024] x [3072,1024]^T, scatter to q/k/v^T --------
__global__ __launch_bounds__(128) void k_qkv_gemm(
    const unsigned short* __restrict__ xb,    // x bf16 [4096][1024]
    const unsigned short* __restrict__ wb,    // w_qkv bf16 [3072][1024]
    const float* __restrict__ bias,           // b_qkv [3072]
    unsigned short* __restrict__ qb,          // [B,H,L,HD] (pre-scaled)
    unsigned short* __restrict__ kb,          // [B,H,L,HD]
    unsigned short* __restrict__ vtb) {       // [B,H,HD,L]
  int wave = (int)((blockIdx.x * blockDim.x + threadIdx.x) >> 5);
  int lane = threadIdx.x & 31;
  const int COLG = THREE_D / 64;              // 48 column groups of 64
  int rt = wave / COLG, cg = wave % COLG;
  int row_base = rt * 16, col_base = cg * 64;
  int nloc = lane & 15, half = lane >> 4;

  v8f acc[4];
#pragma unroll
  for (int t = 0; t < 4; ++t) {
    float bv = bias[col_base + t * 16 + nloc];
#pragma unroll
    for (int i = 0; i < 8; ++i) acc[t][i] = bv;
  }

  gemm_k_pipeline(xb, row_base, wb, col_base, nloc, lane, acc);

#pragma unroll
  for (int t = 0; t < 4; ++t) {
    int col = col_base + t * 16 + nloc;
    int part = col / DD;                      // 0=q 1=k 2=v
    int f = col % DD;
    int h = f / HD, hd = f % HD;
    float scl = (part == 0) ? 0.125f : 1.0f;  // 1/sqrt(HD), folded into q
#pragma unroll
    for (int r = 0; r < 8; ++r) {
      int row = row_base + half * 8 + r;
      int bi = row / LL, tok = row % LL;
      unsigned short val = f2bf_bits(acc[t][r] * scl);
      if (part == 0)
        qb[(((size_t)bi * HH + h) * LL + tok) * HD + hd] = val;
      else if (part == 1)
        kb[(((size_t)bi * HH + h) * LL + tok) * HD + hd] = val;
      else
        vtb[(((size_t)bi * HH + h) * HD + hd) * LL + tok] = val;
    }
  }
}

// ---- Flash attention: one wave per (b, h, 16-query tile) --------------------
__global__ __launch_bounds__(32) void k_attention(
    const unsigned short* __restrict__ qbuf,
    const unsigned short* __restrict__ kbuf,
    const unsigned short* __restrict__ vtbuf,
    const uint32_t* __restrict__ adj,
    unsigned short* __restrict__ attnb) {     // [B, L, D] bf16
  __shared__ unsigned short pT[16 * 48];      // P tile, padded row stride
  int lane = threadIdx.x & 31;
  int gid = blockIdx.x;
  int qt = gid & 63;
  int h = (gid >> 6) & 15;
  int b = gid >> 10;
  int q_base = qt * 16;

  const unsigned short* qh = qbuf + (((size_t)b * HH + h) * LL) * HD;
  const unsigned short* kh = kbuf + (((size_t)b * HH + h) * LL) * HD;
  const unsigned short* vh = vtbuf + (((size_t)b * HH + h) * HD) * LL;

  int nloc = lane & 15, half = lane >> 4;

  v16bf aq0 = load_a16x32(qh, q_base, 0, HD, lane);
  v16bf aq1 = load_a16x32(qh, q_base, 32, HD, lane);

  v8f o[4];
  float mrow[8], lrow[8];
#pragma unroll
  for (int t = 0; t < 4; ++t)
#pragma unroll
    for (int i = 0; i < 8; ++i) o[t][i] = 0.f;
#pragma unroll
  for (int r = 0; r < 8; ++r) { mrow[r] = -3.0e38f; lrow[r] = 0.f; }

  // One 32-key sub-step: scores with bkc, preload bkn for key tile kn,
  // masked online softmax, P@V. Buffers ping-pong across sub-steps.
  auto process = [&](int kv, v16bf* bkc, v16bf* bkn, int kn) {
    // V operands for the CURRENT tile: issue now, consumed after softmax.
    v16bf bv[4];
#pragma unroll
    for (int t4 = 0; t4 < 4; ++t4) {
      const unsigned short* vp = vh + (size_t)(t4 * 16 + nloc) * LL + kv + half * 16;
      bv[t4] = *reinterpret_cast<const v16bf*>(vp);
    }

    // ---- scores S = Q(16x64) @ K^T(64x32): 4 WMMAs ----
    v8f s0, s1;
#pragma unroll
    for (int i = 0; i < 8; ++i) { s0[i] = 0.f; s1[i] = 0.f; }
    s0 = wmma_bf16(aq0, bkc[0], s0);
    s0 = wmma_bf16(aq1, bkc[1], s0);
    s1 = wmma_bf16(aq0, bkc[2], s1);
    s1 = wmma_bf16(aq1, bkc[3], s1);

    // K operands for the NEXT tile: in flight while softmax runs.
    bkn[0] = load_b32x16(kh, kn, 0, HD, lane);
    bkn[1] = load_b32x16(kh, kn, 32, HD, lane);
    bkn[2] = load_b32x16(kh, kn + 16, 0, HD, lane);
    bkn[3] = load_b32x16(kh, kn + 16, 32, HD, lane);

    // WAR fence: previous sub-step's LDS reads must retire before overwrite.
    // (asm memory clobber also pins the loads above before the softmax block)
    asm volatile("s_wait_dscnt 0" ::: "memory");

    // ---- mask + online softmax (fp32) ----
#pragma unroll
    for (int r = 0; r < 8; ++r) {
      int row = q_base + half * 8 + r;
      uint32_t w = adj[(size_t)row * 32 + (kv >> 5)];   // 1 word = 32 keys
      bool m0 = (w >> nloc) & 1u;                        // True = masked (-inf)
      bool m1 = (w >> (nloc + 16)) & 1u;
      float a0 = m0 ? -3.0e38f : s0[r];
      float a1 = m1 ? -3.0e38f : s1[r];
      float t = fmaxf(a0, a1);
      t = fmaxf(t, __shfl_xor(t, 1, 32));
      t = fmaxf(t, __shfl_xor(t, 2, 32));
      t = fmaxf(t, __shfl_xor(t, 4, 32));
      t = fmaxf(t, __shfl_xor(t, 8, 32));
      float mnew = fmaxf(mrow[r], t);
      float mexp = (mnew <= -3.0e38f) ? 0.f : mnew;
      float alpha = (mrow[r] <= -3.0e38f) ? 0.f : __expf(mrow[r] - mnew);
      float p0 = m0 ? 0.f : __expf(a0 - mexp);
      float p1 = m1 ? 0.f : __expf(a1 - mexp);
      float ps = p0 + p1;
      ps += __shfl_xor(ps, 1, 32);
      ps += __shfl_xor(ps, 2, 32);
      ps += __shfl_xor(ps, 4, 32);
      ps += __shfl_xor(ps, 8, 32);
      lrow[r] = lrow[r] * alpha + ps;
      mrow[r] = mnew;
#pragma unroll
      for (int t4 = 0; t4 < 4; ++t4) o[t4][r] *= alpha;
      int prow = half * 8 + r;                // D-layout row -> LDS
      pT[prow * 48 + nloc] = f2bf_bits(p0);
      pT[prow * 48 + nloc + 16] = f2bf_bits(p1);
    }
    // RAW fence: LDS writes visible before A-layout readback.
    asm volatile("s_wait_dscnt 0" ::: "memory");

    // ---- P(16x32) @ V(32x64): 4 WMMAs ----
    V16U ap;
    {
      const unsigned short* p = &pT[(size_t)nloc * 48 + half * 16];
      ap.h[0] = *reinterpret_cast<const v8bf*>(p);
      ap.h[1] = *reinterpret_cast<const v8bf*>(p + 8);
    }
#pragma unroll
    for (int t4 = 0; t4 < 4; ++t4) o[t4] = wmma_bf16(ap.v, bv[t4], o[t4]);
  };

  // Preload K operands for kv=0, then ping-pong bkA/bkB (no copy chain).
  v16bf bkA[4], bkB[4];
  bkA[0] = load_b32x16(kh, 0, 0, HD, lane);
  bkA[1] = load_b32x16(kh, 0, 32, HD, lane);
  bkA[2] = load_b32x16(kh, 16, 0, HD, lane);
  bkA[3] = load_b32x16(kh, 16, 32, HD, lane);

  for (int kv = 0; kv < LL; kv += 64) {
    process(kv, bkA, bkB, kv + 32);
    process(kv + 32, bkB, bkA, (kv + 64) & (LL - 1)); // wraps on last iter
  }

  // ---- normalize + store bf16 [B, L, D] (d = h*64 + hd) ----
#pragma unroll
  for (int r = 0; r < 8; ++r) {
    float inv = (lrow[r] > 0.f) ? 1.0f / lrow[r] : 0.f;
    int row = q_base + half * 8 + r;
#pragma unroll
    for (int t4 = 0; t4 < 4; ++t4) {
      int col = h * HD + t4 * 16 + nloc;
      attnb[((size_t)b * LL + row) * DD + col] = f2bf_bits(o[t4][r] * inv);
    }
  }
}

// ---- Output projection: [4096,1024] x [1024,1024]^T -> fp32 d_out -----------
__global__ __launch_bounds__(128) void k_out_gemm(
    const unsigned short* __restrict__ ab,    // attn bf16 [4096][1024]
    const unsigned short* __restrict__ wb,    // w_out bf16 [1024][1024]
    const float* __restrict__ bias,           // b_out [1024]
    float* __restrict__ out) {                // [4096][1024] fp32
  int wave = (int)((blockIdx.x * blockDim.x + threadIdx.x) >> 5);
  int lane = threadIdx.x & 31;
  const int COLG = DD / 64;                   // 16
  int rt = wave / COLG, cg = wave % COLG;
  int row_base = rt * 16, col_base = cg * 64;
  int nloc = lane & 15, half = lane >> 4;

  v8f acc[4];
#pragma unroll
  for (int t = 0; t < 4; ++t) {
    float bv = bias[col_base + t * 16 + nloc];
#pragma unroll
    for (int i = 0; i < 8; ++i) acc[t][i] = bv;
  }

  gemm_k_pipeline(ab, row_base, wb, col_base, nloc, lane, acc);

#pragma unroll
  for (int t = 0; t < 4; ++t) {
    int col = col_base + t * 16 + nloc;
#pragma unroll
    for (int r = 0; r < 8; ++r) {
      int row = row_base + half * 8 + r;
      out[(size_t)row * DD + col] = acc[t][r];
    }
  }
}

// ---- Host-side orchestration ------------------------------------------------
extern "C" void kernel_launch(void* const* d_in, const int* in_sizes, int n_in,
                              void* d_out, int out_size, void* d_ws, size_t ws_size,
                              hipStream_t stream) {
  const float* x     = (const float*)d_in[0];
  const int*   edges = (const int*)d_in[1];
  const float* w_qkv = (const float*)d_in[2];
  const float* b_qkv = (const float*)d_in[3];
  const float* w_out = (const float*)d_in[4];
  const float* b_out = (const float*)d_in[5];
  float* out = (float*)d_out;
  (void)n_in; (void)out_size; (void)ws_size;

  char* ws = (char*)d_ws;
  size_t off = 0;
  auto carve = [&](size_t bytes) {
    void* p = ws + off;
    off = (off + bytes + 255) & ~(size_t)255;
    return p;
  };
  uint32_t*       adj   = (uint32_t*)      carve((size_t)LL * 32 * 4);        // 128 KB
  unsigned short* xb    = (unsigned short*)carve((size_t)NROWS * DD * 2);     // 8 MB
  unsigned short* wqkvb = (unsigned short*)carve((size_t)THREE_D * DD * 2);   // 6 MB
  unsigned short* woutb = (unsigned short*)carve((size_t)DD * DD * 2);        // 2 MB
  unsigned short* qb    = (unsigned short*)carve((size_t)BB * HH * LL * HD * 2);
  unsigned short* kb    = (unsigned short*)carve((size_t)BB * HH * LL * HD * 2);
  unsigned short* vtb   = (unsigned short*)carve((size_t)BB * HH * LL * HD * 2);
  unsigned short* attnb = (unsigned short*)carve((size_t)NROWS * DD * 2);

  int nAdj = LL * 32;
  k_clear_u32<<<(nAdj + 255) / 256, 256, 0, stream>>>(adj, nAdj);
  int nE = in_sizes[1] / 2;
  k_scatter_edges<<<(nE + 255) / 256, 256, 0, stream>>>(edges, adj, nE);

  int nX = NROWS * DD;
  k_f32_to_bf16<<<(nX + 255) / 256, 256, 0, stream>>>(x, xb, nX);
  int nWq = THREE_D * DD;
  k_f32_to_bf16<<<(nWq + 255) / 256, 256, 0, stream>>>(w_qkv, wqkvb, nWq);
  int nWo = DD * DD;
  k_f32_to_bf16<<<(nWo + 255) / 256, 256, 0, stream>>>(w_out, woutb, nWo);

  // QKV GEMM: 256 row-tiles x 48 col-groups = 12288 waves, 4 waves/block.
  k_qkv_gemm<<<12288 / 4, 128, 0, stream>>>(xb, wqkvb, b_qkv, qb, kb, vtb);

  // Attention: B*H*(L/16) = 4096 single-wave blocks.
  k_attention<<<4096, 32, 0, stream>>>(qb, kb, vtb, adj, attnb);

  // Output GEMM: 256 x 16 = 4096 waves, 4 waves/block.
  k_out_gemm<<<4096 / 4, 128, 0, stream>>>(attnb, woutb, b_out, out);
}
